// MHSA_51960514347454
// MI455X (gfx1250) — matrix operand order repaired and checked by reading, not compiled, not run on hardware
//
#include <hip/hip_runtime.h>

typedef __attribute__((ext_vector_type(16))) __bf16 v16bf;
typedef __attribute__((ext_vector_type(8)))  float  v8f;

union FragBF {
    v16bf v;
    uint4 q[2];
    unsigned short us[16];
};

__device__ __forceinline__ unsigned short f2bf(float f) {
    unsigned int u = __float_as_uint(f);
    unsigned int r = u + 0x7FFFu + ((u >> 16) & 1u);   // round-to-nearest-even
    return (unsigned short)(r >> 16);
}

__device__ __forceinline__ float rmax16(float x) {
    #pragma unroll
    for (int m = 1; m < 16; m <<= 1) x = fmaxf(x, __shfl_xor(x, m, 16));
    return x;
}
__device__ __forceinline__ float rsum16(float x) {
    #pragma unroll
    for (int m = 1; m < 16; m <<= 1) x += __shfl_xor(x, m, 16);
    return x;
}

#define WMMA_BF16(A, B, C) \
    __builtin_amdgcn_wmma_f32_16x16x32_bf16(false, (A), false, (B), (short)0, (C), false, false)

// gfx1250 async copy global->LDS (tracked by ASYNCcnt). GVS mode:
// mem_addr = SGPR64 + VGPR32 + IOFFSET ; LDS byte offset comes from the vdst VGPR.
__device__ __forceinline__ void async_g2l_b128(unsigned lds_off, unsigned gl_off,
                                               unsigned long long base) {
    asm volatile("global_load_async_to_lds_b128 %0, %1, %2"
                 :: "v"(lds_off), "v"(gl_off), "s"(base) : "memory");
}
__device__ __forceinline__ void wait_async0() {
    asm volatile("s_wait_asynccnt 0" ::: "memory");
}

// ---------------------------------------------------------------------------
// Prep: W (512x512 f32) -> bf16
__global__ void mhsa_convW(const float* __restrict__ W, unsigned short* __restrict__ Wbf) {
    int i = blockIdx.x * blockDim.x + threadIdx.x;
    if (i < 512 * 512) Wbf[i] = f2bf(W[i]);
}

// Prep: rel tables (63x128 f32) -> padded (64x128) bf16 (row 63 zero, never read)
__global__ void mhsa_convRel(const float* __restrict__ rw, const float* __restrict__ rh,
                             unsigned short* __restrict__ rwbf, unsigned short* __restrict__ rhbf) {
    int i = blockIdx.x * blockDim.x + threadIdx.x;
    if (i >= 2 * 64 * 128) return;
    int tb = i / (64 * 128);
    int j  = i % (64 * 128);
    int r = j >> 7, d = j & 127;
    const float* src = tb ? rh : rw;
    unsigned short v = (r < 63) ? f2bf(src[r * 128 + d]) : (unsigned short)0;
    (tb ? rhbf : rwbf)[j] = v;
}

// Prep: X[b][c][p] f32 -> Xt[b][p][c] bf16, LDS-tiled transpose. grid(32,16,16) block(32,8)
__global__ void mhsa_tconv(const float* __restrict__ X, unsigned short* __restrict__ Xt) {
    __shared__ float tile[32][33];
    int b  = blockIdx.z;
    int p0 = blockIdx.x * 32, c0 = blockIdx.y * 32;
    int tx = threadIdx.x, ty = threadIdx.y;
    const float* Xb = X + (size_t)b * 512 * 1024;
    #pragma unroll
    for (int i = 0; i < 4; ++i)
        tile[ty + 8 * i][tx] = Xb[(size_t)(c0 + ty + 8 * i) * 1024 + p0 + tx];
    __syncthreads();
    unsigned short* Xtb = Xt + (size_t)b * 1024 * 512;
    #pragma unroll
    for (int i = 0; i < 4; ++i)
        Xtb[(size_t)(p0 + ty + 8 * i) * 512 + c0 + tx] = f2bf(tile[tx][ty + 8 * i]);
}

// ---------------------------------------------------------------------------
// Projection GEMM: C[o][p] = W(512x512) @ X(512x1024), per batch.
// A = Wbf rows (o, c-contig); B = Xt rows (p, c-contig).  One 16x16 tile / wave.
__global__ void mhsa_proj(const unsigned short* __restrict__ Wbf,
                          const unsigned short* __restrict__ Xt,
                          unsigned short* __restrict__ qout,
                          unsigned short* __restrict__ kout,
                          unsigned short* __restrict__ vout,
                          int mode) {
    const int wave = threadIdx.x >> 5, lane = threadIdx.x & 31;
    const int half = lane >> 4, ln = lane & 15;
    const int b = blockIdx.x >> 8;                  // 256 WGs per batch
    const int t = ((blockIdx.x & 255) << 3) + wave; // tile 0..2047
    const int mtile = t >> 6, ntile = t & 63;

    const unsigned short* Xb = Xt + (size_t)b * 1024 * 512;
    const int arow = mtile * 16 + ln;
    const int brow = ntile * 16 + ln;

    v8f c = {};
    #pragma unroll 4
    for (int kk = 0; kk < 16; ++kk) {
        FragBF a, bb;
        const unsigned short* ap = Wbf + (size_t)arow * 512 + kk * 32 + half * 16;
        a.q[0] = *(const uint4*)(ap);
        a.q[1] = *(const uint4*)(ap + 8);
        const unsigned short* bp = Xb + (size_t)brow * 512 + kk * 32 + half * 16;
        bb.q[0] = *(const uint4*)(bp);
        bb.q[1] = *(const uint4*)(bp + 8);
        c = WMMA_BF16(a.v, bb.v, c);
    }

    const int obase = mtile * 16 + half * 8;   // 8 consecutive output channels
    const int head  = obase >> 7;
    const int dbase = obase & 127;
    const int p     = brow;
    const int bn    = b * 4 + head;

    union { unsigned short h[8]; uint4 u; } pk;
    if (mode == 0) {
        const float SCALE = 0.08838834764831845f;   // 128^-0.5
        #pragma unroll
        for (int v = 0; v < 8; ++v) pk.h[v] = f2bf(c[v] * SCALE);
        *(uint4*)(qout + ((size_t)bn * 1024 + p) * 128 + dbase) = pk.u;
    } else {
        #pragma unroll
        for (int v = 0; v < 8; ++v) pk.h[v] = f2bf(c[v]);
        *(uint4*)(kout + ((size_t)bn * 1024 + p) * 128 + dbase) = pk.u;
        #pragma unroll
        for (int v = 0; v < 8; ++v)
            vout[((size_t)bn * 128 + dbase + v) * 1024 + p] = pk.h[v];
    }
}

// ---------------------------------------------------------------------------
// Rel logits: RelW[bn][x][r] = Q[bn][x][:] . rel_w[r][:]  (and RelH), r padded to 64.
__global__ void mhsa_rel(const unsigned short* __restrict__ qbf,
                         const unsigned short* __restrict__ rwbf,
                         const unsigned short* __restrict__ rhbf,
                         float* __restrict__ RelW, float* __restrict__ RelH) {
    const int wave = threadIdx.x >> 5, lane = threadIdx.x & 31;
    const int half = lane >> 4, ln = lane & 15;
    const int bn    = blockIdx.x >> 3;
    const int qtile = ((blockIdx.x & 7) << 3) + wave;   // 0..63
    const int qb    = qtile * 16;

    const unsigned short* Q = qbf + ((size_t)bn * 1024 + qb + ln) * 128;
    FragBF qa[4];
    #pragma unroll
    for (int kk = 0; kk < 4; ++kk) {
        qa[kk].q[0] = *(const uint4*)(Q + kk * 32 + half * 16);
        qa[kk].q[1] = *(const uint4*)(Q + kk * 32 + half * 16 + 8);
    }

    #pragma unroll
    for (int tb = 0; tb < 2; ++tb) {
        const unsigned short* T = tb ? rhbf : rwbf;
        float* Ro = tb ? RelH : RelW;
        #pragma unroll
        for (int nt = 0; nt < 4; ++nt) {
            v8f c = {};
            #pragma unroll
            for (int kk = 0; kk < 4; ++kk) {
                FragBF bb;
                const unsigned short* tr = T + (size_t)(nt * 16 + ln) * 128 + kk * 32 + half * 16;
                bb.q[0] = *(const uint4*)(tr);
                bb.q[1] = *(const uint4*)(tr + 8);
                c = WMMA_BF16(qa[kk].v, bb.v, c);
            }
            #pragma unroll
            for (int v = 0; v < 8; ++v)
                Ro[((size_t)bn * 1024 + qb + v + 8 * half) * 64 + nt * 16 + ln] = c[v];
        }
    }
}

// ---------------------------------------------------------------------------
// Fused flash attention with gathered relative bias.
// grid = 64 bn * 8 qblocks, block = 256 (8 waves); each wave owns 16 queries.
// K/V blocks are staged in LDS once per workgroup via async copies (ASYNCcnt),
// double-buffered so the next block's DMA overlaps the current block's WMMAs.
__global__ void mhsa_attn(const unsigned short* __restrict__ qbf,
                          const unsigned short* __restrict__ kbf,
                          const unsigned short* __restrict__ vbf,
                          const float* __restrict__ RelW,
                          const float* __restrict__ RelH,
                          float* __restrict__ out) {
    __shared__ __align__(16) unsigned short kls[2][32][128];  // K block: 2 x 8KB
    __shared__ __align__(16) unsigned short vls[2][128][32];  // V block: 2 x 8KB
    __shared__ __align__(16) unsigned short pls[8][16][32];   // P reshape per wave

    const int tid  = threadIdx.x;
    const int wave = tid >> 5, lane = tid & 31;
    const int half = lane >> 4, ln = lane & 15;
    const int bn   = blockIdx.x >> 3;
    const int qblk = blockIdx.x & 7;
    const int qimg = qblk * 128 + wave * 16;     // first query pixel of this wave

    // Stage one 32-key block (K: 8KB contiguous, V: 128 rows x 64B) into LDS buf.
    auto stage = [&](int buf, int kb) {
        const unsigned long long kg =
            (unsigned long long)(kbf + ((size_t)bn * 1024 + kb * 32) * 128);
        const unsigned kbase = (unsigned)(size_t)&kls[buf][0][0];
        #pragma unroll
        for (int j = 0; j < 2; ++j) {
            unsigned off = (unsigned)(tid + j * 256) * 16u;    // 0..8191
            async_g2l_b128(kbase + off, off, kg);
        }
        const unsigned long long vg =
            (unsigned long long)(vbf + (size_t)bn * 128 * 1024 + kb * 32);
        #pragma unroll
        for (int j = 0; j < 2; ++j) {
            int c = tid + j * 256;                 // 512 chunks of 16B
            int row = c >> 2, part = c & 3;
            unsigned loff = (unsigned)(size_t)&vls[buf][row][part * 8];
            unsigned goff = (unsigned)((row * 1024 + part * 8) * 2);
            async_g2l_b128(loff, goff, vg);
        }
    };

    // Preload 16x128 Q as 4 A-fragments
    const unsigned short* Q = qbf + ((size_t)bn * 1024 + qimg + ln) * 128;
    FragBF qa[4];
    #pragma unroll
    for (int kk = 0; kk < 4; ++kk) {
        qa[kk].q[0] = *(const uint4*)(Q + kk * 32 + half * 16);
        qa[kk].q[1] = *(const uint4*)(Q + kk * 32 + half * 16 + 8);
    }

    v8f acc[8];
    #pragma unroll
    for (int j = 0; j < 8; ++j) acc[j] = (v8f){};
    float mrow[8], lrow[8];
    #pragma unroll
    for (int v = 0; v < 8; ++v) { mrow[v] = -3.0e38f; lrow[v] = 0.f; }

    stage(0, 0);

    for (int kb = 0; kb < 32; ++kb) {
        const int cur = kb & 1;
        wait_async0();          // my async stores into LDS are done
        __syncthreads();        // => everyone's are done; prior buffer reads done too
        if (kb + 1 < 32) stage(1 - cur, kb + 1);   // overlap next DMA with compute

        // ---- S = Q K^T  (two 16x16 f32 tiles) from LDS ----
        v8f st[2];
        #pragma unroll
        for (int t = 0; t < 2; ++t) {
            v8f c = {};
            #pragma unroll
            for (int kk = 0; kk < 4; ++kk) {
                FragBF bb;
                bb.q[0] = *(const uint4*)(&kls[cur][t * 16 + ln][kk * 32 + half * 16]);
                bb.q[1] = *(const uint4*)(&kls[cur][t * 16 + ln][kk * 32 + half * 16 + 8]);
                c = WMMA_BF16(qa[kk].v, bb.v, c);
            }
            st[t] = c;
        }

        // ---- bias gather + online softmax (rows = M = v + 8*half) ----
        #pragma unroll
        for (int v = 0; v < 8; ++v) {
            const int qoff = qimg + v + 8 * half;      // pixel index of this row
            const int hq = qoff >> 5, wq = qoff & 31;
            const size_t rb = ((size_t)bn * 1024 + qoff) * 64;
            const float bh  = RelH[rb + (kb - hq + 31)];           // hk == kb (aligned block)
            const float bw0 = RelW[rb + (ln - wq + 31)];           // wk = ln
            const float bw1 = RelW[rb + (16 + ln - wq + 31)];      // wk = 16+ln
            float x0 = st[0][v] + bh + bw0;
            float x1 = st[1][v] + bh + bw1;
            float rm = rmax16(fmaxf(x0, x1));
            float mn = fmaxf(mrow[v], rm);
            float corr = __expf(mrow[v] - mn);
            mrow[v] = mn;
            float p0 = __expf(x0 - mn);
            float p1 = __expf(x1 - mn);
            lrow[v] = lrow[v] * corr + rsum16(p0 + p1);
            #pragma unroll
            for (int j = 0; j < 8; ++j) acc[j][v] *= corr;
            pls[wave][v + 8 * half][ln]      = f2bf(p0);
            pls[wave][v + 8 * half][16 + ln] = f2bf(p1);
        }
        asm volatile("" ::: "memory");   // DS ops are in-order per wave

        // Re-shape P (C-layout) -> A-fragment via LDS
        FragBF pf;
    pf.q[0] = *(const uint4*)(&pls[wave][ln][half * 16]);
    pf.q[1] = *(const uint4*)(&pls[wave][ln][half * 16 + 8]);

        // ---- O += P @ V  (V staged [d][pix] => B columns K-contiguous) ----
        #pragma unroll
        for (int j = 0; j < 8; ++j) {
            FragBF vf;
            vf.q[0] = *(const uint4*)(&vls[cur][j * 16 + ln][half * 16]);
            vf.q[1] = *(const uint4*)(&vls[cur][j * 16 + ln][half * 16 + 8]);
            acc[j] = WMMA_BF16(pf.v, vf.v, acc[j]);
        }
    }

    // ---- epilogue: normalize, write out[b][head*128+d][pixel] ----
    float inv[8];
    #pragma unroll
    for (int v = 0; v < 8; ++v) inv[v] = 1.f / lrow[v];
    #pragma unroll
    for (int j = 0; j < 8; ++j) {
        const size_t ch = (size_t)bn * 128 + j * 16 + ln;   // global out channel
        #pragma unroll
        for (int v = 0; v < 8; ++v)
            out[ch * 1024 + (qimg + v + 8 * half)] = acc[j][v] * inv[v];
    }
}

// ---------------------------------------------------------------------------
extern "C" void kernel_launch(void* const* d_in, const int* in_sizes, int n_in,
                              void* d_out, int out_size, void* d_ws, size_t ws_size,
                              hipStream_t stream) {
    const float* qmap = (const float*)d_in[0];   // (16,512,32,32)
    const float* kvmap = (const float*)d_in[1];  // (16,512,32,32)
    const float* W = (const float*)d_in[2];      // (512,512)
    const float* relH = (const float*)d_in[3];   // (63,128)
    const float* relW = (const float*)d_in[4];   // (63,128)
    float* out = (float*)d_out;                  // (16,512,32,32)

    char* ws = (char*)d_ws;
    size_t off = 0;
    auto alloc = [&](size_t bytes) { char* p = ws + off; off += (bytes + 255) & ~(size_t)255; return p; };

    unsigned short* Wbf  = (unsigned short*)alloc(512 * 512 * 2);
    unsigned short* rwbf = (unsigned short*)alloc(64 * 128 * 2);
    unsigned short* rhbf = (unsigned short*)alloc(64 * 128 * 2);
    unsigned short* Xq   = (unsigned short*)alloc((size_t)16 * 1024 * 512 * 2);  // aliased by RelWf later
    unsigned short* Xkv  = (unsigned short*)alloc((size_t)16 * 1024 * 512 * 2);  // aliased by RelHf later
    unsigned short* qbf  = (unsigned short*)alloc((size_t)64 * 1024 * 128 * 2);
    unsigned short* kbf  = (unsigned short*)alloc((size_t)64 * 1024 * 128 * 2);
    unsigned short* vbf  = (unsigned short*)alloc((size_t)64 * 128 * 1024 * 2);
    // Xq / Xkv are dead after the projection GEMMs -> reuse for RelW / RelH (f32, same 16 MB size)
    float* RelWf = (float*)Xq;
    float* RelHf = (float*)Xkv;

    // 1) conversions
    mhsa_convW<<<(512 * 512 + 255) / 256, 256, 0, stream>>>(W, Wbf);
    mhsa_convRel<<<(2 * 64 * 128 + 255) / 256, 256, 0, stream>>>(relW, relH, rwbf, rhbf);
    mhsa_tconv<<<dim3(32, 16, 16), dim3(32, 8), 0, stream>>>(qmap, Xq);
    mhsa_tconv<<<dim3(32, 16, 16), dim3(32, 8), 0, stream>>>(kvmap, Xkv);

    // 2) projections (16 batches * 256 WGs)
    mhsa_proj<<<16 * 256, 256, 0, stream>>>(Wbf, Xq,  qbf, kbf, vbf, 0);
    mhsa_proj<<<16 * 256, 256, 0, stream>>>(Wbf, Xkv, qbf, kbf, vbf, 1);

    // 3) relative logits (64 bn * 8 WGs)  -- writes over Xq/Xkv (now dead)
    mhsa_rel<<<64 * 8, 256, 0, stream>>>(qbf, rwbf, rhbf, RelWf, RelHf);

    // 4) fused flash attention (64 bn * 8 query blocks)
    mhsa_attn<<<64 * 8, 256, 0, stream>>>(qbf, kbf, vbf, RelWf, RelHf, out);
}
